// MLP_small_per_feature_16750372454576
// MI455X (gfx1250) — matrix operand order repaired and checked by reading, not compiled, not run on hardware
//
#include <hip/hip_runtime.h>
#include <hip/hip_bf16.h>

// MI455X / gfx1250: wave32, WMMA 16x16x32 bf16 with f32 accumulate.
typedef __attribute__((ext_vector_type(16))) __bf16 v16bf;
typedef __attribute__((ext_vector_type(8)))  __bf16 v8bf;
typedef __attribute__((ext_vector_type(8)))  float  v8f;

#define F_  256
#define H_  64
#define B_  8192
#define LDW 72          // padded LDS row stride (bf16 elems): 144B -> conflict-free rows
#define BLOCKS_PER_F 8
#define ITERS 8         // 8 chunks * 128 rows * 8 blocks = 8192 rows per feature

__global__ __launch_bounds__(256)
void mlp_per_feature_wmma(const float* __restrict__ x,
                          const float* __restrict__ w1,
                          const float* __restrict__ b1,
                          const float* __restrict__ w2,
                          const float* __restrict__ b2,
                          const float* __restrict__ w3,
                          const float* __restrict__ b3,
                          float* __restrict__ out)
{
    __shared__ __attribute__((aligned(16))) __bf16 w2s[H_ * LDW];      // [out][in], bf16
    __shared__ __attribute__((aligned(16))) __bf16 h1s[8][16 * LDW];   // per-wave A tile
    __shared__ float w1s[H_], b1s[H_], b2s[H_], w3s[H_];

    const int t       = threadIdx.x;
    const int wave    = t >> 5;
    const int lane    = t & 31;
    const int ln      = lane & 15;       // N-lane / row-lane within half
    const int halfSel = lane >> 4;       // 0: lanes 0-15, 1: lanes 16-31
    const int f       = (int)blockIdx.x & (F_ - 1);
    const int blk     = (int)blockIdx.x >> 8;          // 0..7

    // ---- preload per-feature parameters ----
    if (t < H_) {
        w1s[t] = w1[f * H_ + t];
        b1s[t] = b1[f * H_ + t];
        b2s[t] = b2[f * H_ + t];
        w3s[t] = w3[f * H_ + t];
    }
    const float b3f = b3[f];
    for (int i = t; i < H_ * H_; i += 256) {
        const int o = i >> 6, h = i & 63;
        w2s[o * LDW + h] = (__bf16)w2[f * H_ * H_ + i];  // row-major [out][in]
    }
    __syncthreads();

    // ---- hoist the 8 loop-invariant B fragments (w2) into registers ----
    // B is K x N with B[k][n] = w2[n][k]; B layout used here (contiguous k, as A): lane half selects K+16,
    // half j (0..15) -> K = kb + j, N = ln. Contiguous 32B per lane in a w2s row.
    v16bf bfrag[8];
#pragma unroll
    for (int k0i = 0; k0i < 2; ++k0i) {
        const int kb = k0i * 32 + halfSel * 16;
#pragma unroll
        for (int j = 0; j < 4; ++j) {
            const int n = j * 16 + ln;
            const __bf16* p = &w2s[n * LDW + kb];
            v8bf lo = *(const v8bf*)p;
            v8bf hi = *(const v8bf*)(p + 8);
            v16bf bb;
#pragma unroll
            for (int e = 0; e < 8; ++e) { bb[e] = lo[e]; bb[e + 8] = hi[e]; }
            bfrag[k0i * 4 + j] = bb;
        }
    }

    // ---- hoist loop-invariant epilogue scalars (4+4 VGPRs, not 32) ----
    float w3n[4], b2n[4];
#pragma unroll
    for (int j = 0; j < 4; ++j) {
        const int n = j * 16 + ln;
        w3n[j] = w3s[n];
        b2n[j] = b2s[n];
    }

    const int row   = ln;                 // batch row within this wave's 16-row tile
    const int hbase = halfSel * 32;       // layer-1 column half owned by this lane
    __bf16* myTile  = h1s[wave];

    for (int it = 0; it < ITERS; ++it) {
        const int b0 = blk * 1024 + it * 128 + wave * 16;

        // ---- layer 1: h1 = relu(x*w1 + b1), store bf16 tile to LDS ----
        const float xm = x[(size_t)(b0 + row) * F_ + f];
        if (it + 1 < ITERS)
            __builtin_prefetch(&x[(size_t)(b0 + 128 + row) * F_ + f], 0, 0);
#pragma unroll
        for (int tt = 0; tt < 16; ++tt) {
            const int h = hbase + tt * 2;
            const float v0 = fmaxf(fmaf(xm, w1s[h],     b1s[h]),     0.f);
            const float v1 = fmaxf(fmaf(xm, w1s[h + 1], b1s[h + 1]), 0.f);
            union { __bf16 e[2]; unsigned u; } pk;
            pk.e[0] = (__bf16)v0; pk.e[1] = (__bf16)v1;
            *(unsigned*)&myTile[row * LDW + h] = pk.u;
        }
        __builtin_amdgcn_wave_barrier();   // per-wave tile; DS ops are in-order

        // ---- layer 2: [16x64] = [16x64]x[64x64] via 8 WMMAs (C = inline 0) ----
        v8f acc[4] = {};
#pragma unroll
        for (int k0i = 0; k0i < 2; ++k0i) {
            // A 16x32 bf16: lane half -> K+16; halves 0..7 at kb, 8..15 at kb+8
            const int kb = k0i * 32 + halfSel * 16;
            const __bf16* p = &myTile[row * LDW + kb];
            v8bf lo = *(const v8bf*)p;
            v8bf hi = *(const v8bf*)(p + 8);
            v16bf a;
#pragma unroll
            for (int e = 0; e < 8; ++e) { a[e] = lo[e]; a[e + 8] = hi[e]; }
#pragma unroll
            for (int j = 0; j < 4; ++j) {
                acc[j] = __builtin_amdgcn_wmma_f32_16x16x32_bf16(
                    false, a, false, bfrag[k0i * 4 + j],
                    (short)0, acc[j], false, false);
            }
        }

        // ---- layer 3 fused epilogue: out = sum_n relu(acc+b2)*w3[n] + b3 ----
        // The +b2 add makes the relu input known-canonical -> single v_max.
        float part[8];
#pragma unroll
        for (int r = 0; r < 8; ++r) part[r] = 0.f;
#pragma unroll
        for (int j = 0; j < 4; ++j) {
#pragma unroll
            for (int r = 0; r < 8; ++r) {
                const float h2 = fmaxf(acc[j][r] + b2n[j], 0.f);
                part[r] = fmaf(h2, w3n[j], part[r]);
            }
        }

        // ---- reduce-scatter across the 16 lanes of each half (bits 0..3) ----
        // Every lane of a half holds the same 8 m-partials over disjoint n-sets;
        // halve live values each stage: 8 -> 4 -> 2 -> 1.
        float tmp[8];
#pragma unroll
        for (int i = 0; i < 8; ++i) tmp[i] = __shfl_xor(part[i], 1, 32);
        const int o1 = (lane & 1) ? 4 : 0;
        float q[4];
#pragma unroll
        for (int i = 0; i < 4; ++i) q[i] = part[i + o1] + tmp[i + o1];

#pragma unroll
        for (int i = 0; i < 4; ++i) tmp[i] = __shfl_xor(q[i], 2, 32);
        const int o2 = (lane & 2) ? 2 : 0;
        float s0 = q[o2]     + tmp[o2];
        float s1 = q[o2 + 1] + tmp[o2 + 1];

        const float t0 = __shfl_xor(s0, 4, 32);
        const float t1 = __shfl_xor(s1, 4, 32);
        float u = (lane & 4) ? (s1 + t1) : (s0 + t0);

        const float v = u + __shfl_xor(u, 8, 32);

        // lanes with bit3==0 each store one output row:
        // m = 4*bit0 + 2*bit1 + bit2 + 8*halfSel
        if (!(lane & 8)) {
            const int m = ((lane & 1) << 2) | (lane & 2) | ((lane >> 2) & 1)
                        | (halfSel << 3);
            out[(size_t)(b0 + m) * F_ + f] = v + b3f;
        }
    }
}

extern "C" void kernel_launch(void* const* d_in, const int* in_sizes, int n_in,
                              void* d_out, int out_size, void* d_ws, size_t ws_size,
                              hipStream_t stream)
{
    (void)in_sizes; (void)n_in; (void)out_size; (void)d_ws; (void)ws_size;
    const float* x  = (const float*)d_in[0];
    const float* w1 = (const float*)d_in[1];
    const float* b1 = (const float*)d_in[2];
    const float* w2 = (const float*)d_in[3];
    const float* b2 = (const float*)d_in[4];
    const float* w3 = (const float*)d_in[5];
    const float* b3 = (const float*)d_in[6];
    float* out = (float*)d_out;

    dim3 grid(F_ * BLOCKS_PER_F);   // 2048 blocks
    dim3 block(256);                // 8 wave32s
    hipLaunchKernelGGL(mlp_per_feature_wmma, grid, block, 0, stream,
                       x, w1, b1, w2, b2, w3, b3, out);
}
